// RNNRegressor_40647570489772
// MI455X (gfx1250) — matrix operand order repaired and checked by reading, not compiled, not run on hardware
//
#include <hip/hip_runtime.h>

// ---------------------------------------------------------------------------
// Fused 2-layer GRU + FC head for gfx1250 (MI455X), wave32 + WMMA bf16.
//   B=2048, S=365, F=64, H=200.
// Each workgroup (512 threads = 16 waves) owns a 32-row batch tile (two WMMA
// M tiles) and runs the entire 365-step recurrence for both GRU layers with
// h kept on-chip.  Each B-fragment (L2-resident packed bf16 weights) is
// loaded once and feeds TWO v_wmma_f32_16x16x32_bf16 ops (independent
// accumulators), doubling FLOP per L2 byte vs a 16-row tile.
// r/z gates are stored as fused xg+hg+bias sums; n-gate operands separately.
// fp32 master h lives in per-thread registers (13 elements/thread).
// ---------------------------------------------------------------------------

typedef __attribute__((ext_vector_type(16))) __bf16 v16bf;
typedef __attribute__((ext_vector_type(8)))  __bf16 v8bf;
typedef __attribute__((ext_vector_type(8)))  float  v8f;

union ABu { v16bf v; struct { v8bf lo, hi; } p; };

#define SEQ      365
#define FEAT     64
#define HID      200
#define BT       32          // batch rows per workgroup (two WMMA M tiles)
#define HP       224         // K padded to 7 chunks of 32
#define GP       208         // per-gate N padded to 13 tiles of 16
#define RZW      (2*GP)      // r/z sum buffer row stride (416)
#define NTILES   39          // 3 gates * 13 N-tiles
#define TPG      13
#define NWAVES   16
#define NTHREADS 512
#define HITER    13          // ceil(BT*HID / NTHREADS)

// ---------------------------------------------------------------------------
// Pack one weight matrix (3H x K, row-major fp32) into bf16 WMMA B fragments.
// Fragment (tIdx, kc): 32 lanes x 16 halves.  Lane L holds column
// n = tile*16 + (L&15), K-range kc*32 + (L>>4)*16 .. +15 (contiguous).
// Zero-padded for n_local >= 200 or k >= K.
// ---------------------------------------------------------------------------
__global__ void pack_weights_kernel(__bf16* __restrict__ dst,
                                    const float* __restrict__ W,
                                    int K, int KCH) {
  int flat = blockIdx.x * blockDim.x + threadIdx.x;
  int total = NTILES * KCH * 512;
  if (flat >= total) return;
  int h    = flat & 15;
  int L    = (flat >> 4) & 31;
  int rest = flat >> 9;               // tIdx*KCH + kc
  int kc   = rest % KCH;
  int tIdx = rest / KCH;
  int gate = tIdx / TPG;
  int nt   = tIdx - gate * TPG;
  int n_local = nt * 16 + (L & 15);
  int k       = kc * 32 + ((L >> 4) << 4) + h;
  float val = (n_local < HID && k < K) ? W[(size_t)(gate * HID + n_local) * K + k]
                                       : 0.0f;
  dst[flat] = (__bf16)val;
}

// Fold the FC head:  out[b] = sum_i last[b][i] * v[i] + c
__global__ void fc_fold_kernel(float* __restrict__ v, float* __restrict__ c,
                               const float* __restrict__ Wfc1,
                               const float* __restrict__ bfc1,
                               const float* __restrict__ Wfc2,
                               const float* __restrict__ bfc2) {
  int i = blockIdx.x * blockDim.x + threadIdx.x;
  if (i < HID) {
    float a = 0.0f;
    for (int j = 0; j < HID; ++j) a += Wfc2[j] * Wfc1[(size_t)j * HID + i];
    v[i] = a;
  } else if (i == HID) {
    float a = bfc2[0];
    for (int j = 0; j < HID; ++j) a += Wfc2[j] * bfc1[j];
    *c = a;
  }
}

// ---------------------------------------------------------------------------
// One GRU layer's projections for this wave's N-tiles.
//   X-side:  A = sAX (32 x KCHX*32 bf16, stride axStride halves), B = fragX
//   H-side:  A = sAH (32 x HP bf16),                              B = fragH
// Each B fragment feeds two WMMAs (M rows 0-15 and 16-31).
// Epilogue folds biases and writes: r/z -> fused sums, n -> xn / hn.
// ---------------------------------------------------------------------------
template <int KCHX>
__device__ __forceinline__ void layer_gemm(
    const __bf16* __restrict__ sAX, int axStride,
    const __bf16* __restrict__ sAH,
    const __bf16* __restrict__ fragX, const __bf16* __restrict__ fragH,
    const float* __restrict__ bih, const float* __restrict__ bhh,
    float* __restrict__ s_rz, float* __restrict__ s_xn,
    float* __restrict__ s_hn, int wv, int lane) {
  const int mrow  = lane & 15;
  const int kh    = (lane >> 4) << 4;   // A-operand K sub-offset (0 or 16)
  const int mBase = (lane >> 4) << 3;   // D-operand row base (0 or 8)
  const int nlane = lane & 15;
  for (int tIdx = wv; tIdx < NTILES; tIdx += NWAVES) {
    v8f aX0 = {}, aX1 = {}, aH0 = {}, aH1 = {};
    const __bf16* fbX = fragX + (size_t)tIdx * KCHX * 512 + lane * 16;
    const __bf16* fbH = fragH + (size_t)tIdx * 7 * 512 + lane * 16;
#pragma unroll
    for (int kc = 0; kc < KCHX; ++kc) {
      ABu a0, a1, b;
      const __bf16* ap = sAX + mrow * axStride + kc * 32 + kh;
      a0.p.lo = *(const v8bf*)(ap);
      a0.p.hi = *(const v8bf*)(ap + 8);
      const __bf16* ap1 = ap + 16 * axStride;
      a1.p.lo = *(const v8bf*)(ap1);
      a1.p.hi = *(const v8bf*)(ap1 + 8);
      const __bf16* bp = fbX + kc * 512;
      b.p.lo = *(const v8bf*)(bp);
      b.p.hi = *(const v8bf*)(bp + 8);
      aX0 = __builtin_amdgcn_wmma_f32_16x16x32_bf16(
          false, a0.v, false, b.v, (short)0, aX0, false, false);
      aX1 = __builtin_amdgcn_wmma_f32_16x16x32_bf16(
          false, a1.v, false, b.v, (short)0, aX1, false, false);
    }
#pragma unroll
    for (int kc = 0; kc < 7; ++kc) {
      ABu a0, a1, b;
      const __bf16* ap = sAH + mrow * HP + kc * 32 + kh;
      a0.p.lo = *(const v8bf*)(ap);
      a0.p.hi = *(const v8bf*)(ap + 8);
      const __bf16* ap1 = ap + 16 * HP;
      a1.p.lo = *(const v8bf*)(ap1);
      a1.p.hi = *(const v8bf*)(ap1 + 8);
      const __bf16* bp = fbH + kc * 512;
      b.p.lo = *(const v8bf*)(bp);
      b.p.hi = *(const v8bf*)(bp + 8);
      aH0 = __builtin_amdgcn_wmma_f32_16x16x32_bf16(
          false, a0.v, false, b.v, (short)0, aH0, false, false);
      aH1 = __builtin_amdgcn_wmma_f32_16x16x32_bf16(
          false, a1.v, false, b.v, (short)0, aH1, false, false);
    }
    int gate = tIdx / TPG;
    int nt   = tIdx - gate * TPG;
    int ncol = nt * 16 + nlane;
    int bcol = gate * HID + (ncol < HID ? ncol : HID - 1);
    float bX = bih[bcol], bH = bhh[bcol];
    if (gate < 2) {                       // r / z : fused pre-sigmoid sums
      float* dst = s_rz + gate * GP + ncol;
      float bsum = bX + bH;
#pragma unroll
      for (int r = 0; r < 8; ++r) {
        dst[(mBase + r) * RZW]        = aX0[r] + aH0[r] + bsum;
        dst[(mBase + r + 16) * RZW]   = aX1[r] + aH1[r] + bsum;
      }
    } else {                              // n : keep xn and hn separate
      float* dx = s_xn + ncol;
      float* dh = s_hn + ncol;
#pragma unroll
      for (int r = 0; r < 8; ++r) {
        dx[(mBase + r) * GP]      = aX0[r] + bX;
        dh[(mBase + r) * GP]      = aH0[r] + bH;
        dx[(mBase + r + 16) * GP] = aX1[r] + bX;
        dh[(mBase + r + 16) * GP] = aH1[r] + bH;
      }
    }
  }
}

// Elementwise GRU gate math.  fp32 master h lives in per-thread registers
// (hreg[i] holds element idx = tid + i*NTHREADS of the 32x200 tile).
__device__ __forceinline__ void gate_math(const float* __restrict__ s_rz,
                                          const float* __restrict__ s_xn,
                                          const float* __restrict__ s_hn,
                                          float* __restrict__ hreg,
                                          __bf16* __restrict__ hbf, int tid) {
#pragma unroll
  for (int i = 0; i < HITER; ++i) {
    int idx = tid + i * NTHREADS;
    if (idx < BT * HID) {
      int m = idx / HID;
      int j = idx - m * HID;
      float r = 1.0f / (1.0f + __expf(-s_rz[m * RZW + j]));
      float z = 1.0f / (1.0f + __expf(-s_rz[m * RZW + GP + j]));
      float n = tanhf(s_xn[m * GP + j] + r * s_hn[m * GP + j]);
      float hnew = (1.0f - z) * n + z * hreg[i];
      hreg[i] = hnew;
      hbf[m * HP + j] = (__bf16)hnew;
    }
  }
}

__global__ __launch_bounds__(NTHREADS) void gru_fused_kernel(
    const float* __restrict__ X,
    const float* __restrict__ bih0, const float* __restrict__ bhh0,
    const float* __restrict__ bih1, const float* __restrict__ bhh1,
    const __bf16* __restrict__ fWih0, const __bf16* __restrict__ fWhh0,
    const __bf16* __restrict__ fWih1, const __bf16* __restrict__ fWhh1,
    const float* __restrict__ vfc, const float* __restrict__ cfc,
    float* __restrict__ out) {
  __shared__ __align__(16) __bf16 s_h0[BT * HP];   // layer0 h (bf16, K-padded)
  __shared__ __align__(16) __bf16 s_h1[BT * HP];   // layer1 h
  __shared__ __align__(16) __bf16 s_x[BT * FEAT];  // X tile (bf16)
  __shared__ float s_rz[BT * RZW];                 // fused r/z pre-sigmoid sums
  __shared__ float s_xn[BT * GP];                  // n-gate input projection
  __shared__ float s_hn[BT * GP];                  // n-gate hidden projection
  __shared__ float s_b[4 * 3 * HID];               // staged biases
  // total static LDS = 14336*2 + 4096 + 53248 + 26624*2 + 9600 = 148,864 B

  const int tid  = threadIdx.x;
  const int lane = tid & 31;
  const int wv   = tid >> 5;
  const int b0   = blockIdx.x * BT;

  float h0f[HITER], h1f[HITER];
#pragma unroll
  for (int i = 0; i < HITER; ++i) { h0f[i] = 0.0f; h1f[i] = 0.0f; }

  for (int i = tid; i < BT * HP; i += NTHREADS) {
    s_h0[i] = (__bf16)0.0f;
    s_h1[i] = (__bf16)0.0f;
  }
  for (int i = tid; i < 3 * HID; i += NTHREADS) {
    s_b[i]           = bih0[i];
    s_b[3 * HID + i] = bhh0[i];
    s_b[6 * HID + i] = bih1[i];
    s_b[9 * HID + i] = bhh1[i];
  }
  __syncthreads();

  for (int t = 0; t < SEQ; ++t) {
    // ---- stage X[:, t, :] tile into LDS as bf16 (32 rows x 64 feats) ----
    {
      int m = tid >> 4, f4 = tid & 15;
      const float4 xv =
          *(const float4*)(X + ((size_t)(b0 + m) * SEQ + t) * FEAT + f4 * 4);
      __bf16* dst = s_x + m * FEAT + f4 * 4;
      dst[0] = (__bf16)xv.x;
      dst[1] = (__bf16)xv.y;
      dst[2] = (__bf16)xv.z;
      dst[3] = (__bf16)xv.w;
    }
    __syncthreads();

    // ---- layer 0: xg = x @ Wih0^T (K=64), hg = h0 @ Whh0^T (K=224) ----
    layer_gemm<2>(s_x, FEAT, s_h0, fWih0, fWhh0, s_b, s_b + 3 * HID, s_rz,
                  s_xn, s_hn, wv, lane);
    __syncthreads();
    gate_math(s_rz, s_xn, s_hn, h0f, s_h0, tid);
    __syncthreads();

    // ---- layer 1: xg = h0 @ Wih1^T, hg = h1 @ Whh1^T ----
    layer_gemm<7>(s_h0, HP, s_h1, fWih1, fWhh1, s_b + 6 * HID, s_b + 9 * HID,
                  s_rz, s_xn, s_hn, wv, lane);
    __syncthreads();
    gate_math(s_rz, s_xn, s_hn, h1f, s_h1, tid);
    __syncthreads();
  }

  // ---- folded FC head on final h1: dump register h1 to LDS, then dot ----
#pragma unroll
  for (int i = 0; i < HITER; ++i) {
    int idx = tid + i * NTHREADS;
    if (idx < BT * HID) s_rz[idx] = h1f[i];   // flat [m*200 + j]
  }
  __syncthreads();
  if (tid < BT) {
    float acc = cfc[0];
    for (int j = 0; j < HID; ++j) acc += s_rz[tid * HID + j] * vfc[j];
    out[b0 + tid] = acc;
  }
}

// ---------------------------------------------------------------------------
// Workspace layout (bytes):
//   [0,800)          v (200 f32)
//   [800,804)        c (1 f32)
//   [1024, +79872)   fragWih0  (39 tiles * 2 kchunks * 1024B)
//   then 3x 279552B  fragWhh0, fragWih1, fragWhh1      (total ~0.92 MB)
// ---------------------------------------------------------------------------
extern "C" void kernel_launch(void* const* d_in, const int* in_sizes, int n_in,
                              void* d_out, int out_size, void* d_ws,
                              size_t ws_size, hipStream_t stream) {
  const float* X    = (const float*)d_in[0];
  const float* Wih0 = (const float*)d_in[1];
  const float* Whh0 = (const float*)d_in[2];
  const float* bih0 = (const float*)d_in[3];
  const float* bhh0 = (const float*)d_in[4];
  const float* Wih1 = (const float*)d_in[5];
  const float* Whh1 = (const float*)d_in[6];
  const float* bih1 = (const float*)d_in[7];
  const float* bhh1 = (const float*)d_in[8];
  const float* Wfc1 = (const float*)d_in[9];
  const float* bfc1 = (const float*)d_in[10];
  const float* Wfc2 = (const float*)d_in[11];
  const float* bfc2 = (const float*)d_in[12];

  char* ws = (char*)d_ws;
  float* vfc = (float*)ws;
  float* cfc = (float*)(ws + 800);
  const size_t FR_IH0 = 1024;
  const size_t SZ_IH0 = (size_t)NTILES * 2 * 1024;   // 79872
  const size_t SZ_HH  = (size_t)NTILES * 7 * 1024;   // 279552
  __bf16* fWih0 = (__bf16*)(ws + FR_IH0);
  __bf16* fWhh0 = (__bf16*)(ws + FR_IH0 + SZ_IH0);
  __bf16* fWih1 = (__bf16*)(ws + FR_IH0 + SZ_IH0 + SZ_HH);
  __bf16* fWhh1 = (__bf16*)(ws + FR_IH0 + SZ_IH0 + 2 * SZ_HH);

  // Pack weights into WMMA fragment layout (deterministic, every launch).
  {
    int total2 = NTILES * 2 * 512;
    int total7 = NTILES * 7 * 512;
    pack_weights_kernel<<<(total2 + 255) / 256, 256, 0, stream>>>(fWih0, Wih0,
                                                                  FEAT, 2);
    pack_weights_kernel<<<(total7 + 255) / 256, 256, 0, stream>>>(fWhh0, Whh0,
                                                                  HID, 7);
    pack_weights_kernel<<<(total7 + 255) / 256, 256, 0, stream>>>(fWih1, Wih1,
                                                                  HID, 7);
    pack_weights_kernel<<<(total7 + 255) / 256, 256, 0, stream>>>(fWhh1, Whh1,
                                                                  HID, 7);
    fc_fold_kernel<<<1, 256, 0, stream>>>(vfc, cfc, Wfc1, bfc1, Wfc2, bfc2);
  }

  // Fused persistent GRU: 2048/32 = 64 workgroups, 512 threads each.
  gru_fused_kernel<<<64, NTHREADS, 0, stream>>>(X, bih0, bhh0, bih1, bhh1,
                                                fWih0, fWhh0, fWih1, fWhh1,
                                                vfc, cfc, (float*)d_out);
}
